// DimeNetPlusPlusGraph_36352603194135
// MI455X (gfx1250) — matrix operand order repaired and checked by reading, not compiled, not run on hardware
//
#include <hip/hip_runtime.h>
#include <math.h>

// ---------------------------------------------------------------------------
// DimeNet++-style graph network for MI455X (gfx1250).
// All dense matmuls run through v_wmma_f32_16x16x32_bf16 (fp32 accumulate,
// bf16 operands staged in LDS, native v_cvt f32->bf16 conversion).
// Activations stay fp32 in HBM (memory-bound workload @ 23.3 TB/s).
// Tile staging uses unpredicated vectorized global_load_b64 + packed
// ds_store_b32 on interior tiles; scalar guarded path only on edge tiles.
// Irregular ops (segment sums, GAT softmax, triplet scatter) use fp32
// global atomics.
// ---------------------------------------------------------------------------

#define NNODES 20000
#define NEDGES 300000
#define NTRIP  1000000
#define HIDD   128
#define INTE   64
#define OUTE   256
#define NRADD  6
#define NSR    42       // NSPH*NRAD
#define HEADSN 4
#define BONDD  64
#define NBLK   3
#define RBFCATW (NRADD + BONDD)   // 70
#define CUTOFFF 5.0f

typedef __attribute__((ext_vector_type(16))) __bf16 v16bf;
typedef __attribute__((ext_vector_type(8)))  float  v8f;

__device__ __forceinline__ float swishf(float x) { return x / (1.0f + __expf(-x)); }

// native fptrunc -> v_cvt (RNE) instead of manual integer rounding
__device__ __forceinline__ unsigned short f2bf(float x) {
  __bf16 b = (__bf16)x;
  union { __bf16 b; unsigned short u; } cv;
  cv.b = b;
  return cv.u;
}
__device__ __forceinline__ unsigned int f2bf2(float lo, float hi) {
  union { __bf16 b[2]; unsigned int u; } cv;
  cv.b[0] = (__bf16)lo;
  cv.b[1] = (__bf16)hi;
  return cv.u;
}

__device__ __forceinline__ void atomicMaxF(float* addr, float val) {
  unsigned int* ua = (unsigned int*)addr;
  unsigned int old = *ua;
  while (__uint_as_float(old) < val) {
    unsigned int assumed = old;
    old = atomicCAS(ua, assumed, __float_as_uint(val));
    if (old == assumed) break;
  }
}

// ---------------------------------------------------------------------------
// Generic WMMA GEMM:  C[M,N] = addsrc + act(A[M,K] @ W[K,N] + bias)
// block tile 128 x BNT, 8 waves (4x2), wave tile 32 x (BNT/2),
// i.e. 2 x (BNT/32) WMMA 16x16x32 bf16 per k-step per wave.
// ---------------------------------------------------------------------------
#define BM 128
#define BK 32
#define LDT (BK + 4)   // halfword row stride (72B): 8B-aligned rows,
                       // 18-u32 stride -> conflict-free fragment reads

template <int BNT>
__global__ void __launch_bounds__(256)
k_gemm(const float* __restrict__ A, const float* __restrict__ W,
       const float* __restrict__ bias, const float* __restrict__ addsrc,
       float* __restrict__ C, int M, int N, int K, int act)
{
  constexpr int NS = BNT / 32;          // n-subtiles per wave (2 or 4)
  __shared__ unsigned short As[BM * LDT];
  __shared__ unsigned short Bs[BNT * LDT];

  const int tid  = threadIdx.x;
  const int lane = tid & 31;
  const int wid  = tid >> 5;
  const int wm   = (wid >> 1) * 32;          // wave row offset in tile
  const int wn   = (wid & 1) * (BNT / 2);    // wave col offset in tile
  const int m16  = lane & 15;
  const int kh   = lane >> 4;                // K-half select per ISA layout
  const int row0 = blockIdx.x * BM;
  const int col0 = blockIdx.y * BNT;

  v8f acc[2][NS];
#pragma unroll
  for (int a = 0; a < 2; ++a)
#pragma unroll
    for (int b = 0; b < NS; ++b) {
      v8f z = {0.f,0.f,0.f,0.f,0.f,0.f,0.f,0.f};
      acc[a][b] = z;
    }

  for (int k0 = 0; k0 < K; k0 += BK) {
    // ---- stage A tile (BM x BK): fp32 -> bf16 ---------------------------
    if (row0 + BM <= M && k0 + BK <= K) {
      // interior tile: coalesced float2 loads, packed b32 LDS stores
#pragma unroll
      for (int i = 0; i < (BM * BK / 2) / 256; ++i) {   // 8 iters
        int idx = tid + i * 256;
        int r  = idx >> 4;                 // 16 float2 per 32-wide row
        int c2 = (idx & 15) * 2;
        const float2 v =
            *(const float2*)(A + (long)(row0 + r) * K + k0 + c2);
        *(unsigned int*)&As[r * LDT + c2] = f2bf2(v.x, v.y);
      }
    } else {
      for (int idx = tid; idx < BM * BK; idx += 256) {
        int r = idx >> 5, c = idx & 31;
        int gr = row0 + r, gk = k0 + c;
        float v = (gr < M && gk < K) ? A[(long)gr * K + gk] : 0.0f;
        As[r * LDT + c] = f2bf(v);
      }
    }
    // ---- stage B tile transposed ([n][k]) -------------------------------
    if (col0 + BNT <= N && k0 + BK <= K) {
#pragma unroll
      for (int i = 0; i < (BNT * BK / 2) / 256; ++i) {  // 4 or 8 iters
        int idx = tid + i * 256;
        int n2 = (idx % (BNT / 2)) * 2;    // consecutive lanes -> consec n
        int c  = idx / (BNT / 2);
        const float2 v =
            *(const float2*)(W + (long)(k0 + c) * N + col0 + n2);
        Bs[n2 * LDT + c]       = f2bf(v.x);
        Bs[(n2 + 1) * LDT + c] = f2bf(v.y);
      }
    } else {
      for (int idx = tid; idx < BNT * BK; idx += 256) {
        int n = idx % BNT, c = idx / BNT;
        int gn = col0 + n, gk = k0 + c;
        float v = (gn < N && gk < K) ? W[(long)gk * N + gn] : 0.0f;
        Bs[n * LDT + c] = f2bf(v);
      }
    }
    __syncthreads();

    if (k0 + BK < K) {                       // global_prefetch_b8 next A tile
      int pr = row0 + (tid & 127);
      if (pr < M) __builtin_prefetch(&A[(long)pr * K + k0 + BK], 0, 0);
    }

    union U { unsigned int u[8]; v16bf v; };
    U af[2], bfv[NS];
#pragma unroll
    for (int t = 0; t < 2; ++t) {
      // A 16-bit 16x32 lane layout: row=m16, K in {kh*8..+7, 16+kh*8..+7}
      const unsigned int* ap =
          (const unsigned int*)&As[(wm + 16 * t + m16) * LDT];
#pragma unroll
      for (int jj = 0; jj < 4; ++jj) {
        af[t].u[jj]     = ap[kh * 4 + jj];
        af[t].u[4 + jj] = ap[8 + kh * 4 + jj];
      }
    }
#pragma unroll
    for (int t = 0; t < NS; ++t) {
      // B 32x16 lane layout: col=m16, K in {kh*8..+7, 16+kh*8..+7}
      const unsigned int* bp =
          (const unsigned int*)&Bs[(wn + 16 * t + m16) * LDT];
#pragma unroll
      for (int jj = 0; jj < 8; ++jj) bfv[t].u[jj] = bp[(jj >> 2) * 8 + kh * 4 + (jj & 3)];
    }

#pragma unroll
    for (int ti = 0; ti < 2; ++ti)
#pragma unroll
      for (int tj = 0; tj < NS; ++tj)
        acc[ti][tj] = __builtin_amdgcn_wmma_f32_16x16x32_bf16(
            false, af[ti].v, false, bfv[tj].v, (short)0, acc[ti][tj],
            false, false);
    __syncthreads();
  }

  __builtin_amdgcn_s_wait_tensorcnt(0);

  // epilogue: C/D layout — VGPR r holds M = r + 8*kh, N = m16
#pragma unroll
  for (int ti = 0; ti < 2; ++ti) {
#pragma unroll
    for (int tj = 0; tj < NS; ++tj) {
      int gc = col0 + wn + 16 * tj + m16;
#pragma unroll
      for (int r = 0; r < 8; ++r) {
        int gr = row0 + wm + 16 * ti + kh * 8 + r;
        if (gr < M && gc < N) {
          float v = acc[ti][tj][r];
          if (bias)   v += bias[gc];
          if (act)    v = swishf(v);
          if (addsrc) v += addsrc[(long)gr * N + gc];
          C[(long)gr * N + gc] = v;
        }
      }
    }
  }
}

// ---------------------------------------------------------------------------
// elementwise / gather / scatter kernels
// ---------------------------------------------------------------------------
__global__ void k_zero(float* p, long n) {
  long i = (long)blockIdx.x * blockDim.x + threadIdx.x;
  if (i < n) p[i] = 0.0f;
}
__global__ void k_fill(float* p, float v, long n) {
  long i = (long)blockIdx.x * blockDim.x + threadIdx.x;
  if (i < n) p[i] = v;
}
__global__ void k_copy(float* d, const float* s, long n) {
  long i = (long)blockIdx.x * blockDim.x + threadIdx.x;
  if (i < n) d[i] = s[i];
}
__global__ void k_mul(float* d, const float* a, const float* b, long n) {
  long i = (long)blockIdx.x * blockDim.x + threadIdx.x;
  if (i < n) d[i] = a[i] * b[i];
}

__global__ void k_node_emb(float* emb, const int* x,
                           const float* resi, const float* atom) {
  long i = (long)blockIdx.x * blockDim.x + threadIdx.x;
  if (i >= (long)NNODES * HIDD) return;
  int n = (int)(i / HIDD), c = (int)(i % HIDD);
  emb[i] = (c < 64) ? resi[x[n * 2 + 0] * 64 + c]
                    : atom[x[n * 2 + 1] * 64 + (c - 64)];
}

__global__ void k_rbf(float* rbf, const float* dist, const float* freq) {
  long i = (long)blockIdx.x * blockDim.x + threadIdx.x;
  if (i >= (long)NEDGES * NRADD) return;
  int e = (int)(i / NRADD), k = (int)(i % NRADD);
  float d = dist[e] * (1.0f / CUTOFFF);
  float d2 = d * d, d4 = d2 * d2, d5 = d4 * d;
  // envelope p=6: 1/r - 28 r^5 + 48 r^6 - 21 r^7
  float env = 1.0f / d - 28.0f * d5 + 48.0f * d5 * d - 21.0f * d5 * d2;
  rbf[i] = env * __sinf(freq[k] * d);
}

// cat[e, 0:128]=feat[i], [128:256]=feat[j], [256:384]=swish(rbf@Wr + br)
__global__ void k_cat3(float* cat, const float* feat, const int* eidx,
                       const float* rbf, const float* Wr, const float* br) {
  long idx = (long)blockIdx.x * blockDim.x + threadIdx.x;
  if (idx >= (long)NEDGES * 384) return;
  int e = (int)(idx / 384), c = (int)(idx % 384);
  float v;
  if (c < 128) {
    v = feat[(long)eidx[NEDGES + e] * HIDD + c];          // dst i
  } else if (c < 256) {
    v = feat[(long)eidx[e] * HIDD + (c - 128)];           // src j
  } else {
    int cc = c - 256;
    float s = br[cc];
    const float* re = rbf + (long)e * NRADD;
#pragma unroll
    for (int k = 0; k < NRADD; ++k) s += re[k] * Wr[k * HIDD + cc];
    v = swishf(s);
  }
  cat[idx] = v;
}

__global__ void k_gat_coef(float* a_s, float* a_d, const float* wx,
                           const float* asrc, const float* adst) {
  int i = blockIdx.x * blockDim.x + threadIdx.x;
  if (i >= NNODES * HEADSN) return;
  int n = i / HEADSN, h = i % HEADSN;
  float ss = 0.f, sd = 0.f;
  const float* w = wx + (long)n * 512 + h * HIDD;
  for (int f = 0; f < HIDD; ++f) {
    ss += w[f] * asrc[h * HIDD + f];
    sd += w[f] * adst[h * HIDD + f];
  }
  a_s[i] = ss; a_d[i] = sd;
}

__global__ void k_att1(float* eatt, float* mbuf, const float* a_s,
                       const float* a_d, const int* eidx) {
  long idx = (long)blockIdx.x * blockDim.x + threadIdx.x;
  if (idx >= (long)(NEDGES + NNODES) * HEADSN) return;
  int ee = (int)(idx / HEADSN), h = (int)(idx % HEADSN);
  int s = (ee < NEDGES) ? eidx[ee] : (ee - NEDGES);
  int d = (ee < NEDGES) ? eidx[NEDGES + ee] : (ee - NEDGES);
  float v = a_s[s * HEADSN + h] + a_d[d * HEADSN + h];
  if (v < 0.f) v *= 0.2f;                              // leaky_relu(0.2)
  eatt[idx] = v;
  atomicMaxF(&mbuf[d * HEADSN + h], v);
}

__global__ void k_att2(float* eatt, float* den, const float* mbuf,
                       const int* eidx) {
  long idx = (long)blockIdx.x * blockDim.x + threadIdx.x;
  if (idx >= (long)(NEDGES + NNODES) * HEADSN) return;
  int ee = (int)(idx / HEADSN), h = (int)(idx % HEADSN);
  int d = (ee < NEDGES) ? eidx[NEDGES + ee] : (ee - NEDGES);
  float ex = __expf(eatt[idx] - mbuf[d * HEADSN + h]);
  eatt[idx] = ex;
  atomicAdd(&den[d * HEADSN + h], ex);
}

__global__ void k_att3(float* gout, const float* eatt, const float* den,
                       const float* wx, const int* eidx) {
  long idx = (long)blockIdx.x * blockDim.x + threadIdx.x;
  if (idx >= (long)(NEDGES + NNODES) * 512) return;
  int ee = (int)(idx / 512), c = (int)(idx % 512);
  int h = c / HIDD;
  int s = (ee < NEDGES) ? eidx[ee] : (ee - NEDGES);
  int d = (ee < NEDGES) ? eidx[NEDGES + ee] : (ee - NEDGES);
  float alpha = eatt[(long)ee * HEADSN + h] /
                (den[d * HEADSN + h] + 1e-16f);
  atomicAdd(&gout[(long)d * 512 + c], alpha * wx[(long)s * 512 + c]);
}

__global__ void k_inorm(float* xn, const float* gout, const float* gat_b,
                        const float* emb) {
  int i = blockIdx.x * blockDim.x + threadIdx.x;
  if (i >= NNODES * HEADSN) return;
  int n = i / HEADSN, h = i % HEADSN;
  const float* g = gout + (long)n * 512 + h * HIDD;
  const float* b = gat_b + h * HIDD;
  const float* e = emb + (long)n * HIDD;
  float mu = 0.f;
  for (int f = 0; f < HIDD; ++f) mu += g[f] + b[f] + e[f];
  mu *= (1.0f / HIDD);
  float var = 0.f;
  for (int f = 0; f < HIDD; ++f) {
    float t = g[f] + b[f] + e[f] - mu;
    var += t * t;
  }
  float inv = rsqrtf(var * (1.0f / HIDD) + 1e-5f);
  float* o = xn + (long)n * 512 + h * HIDD;
  for (int f = 0; f < HIDD; ++f) o[f] = (g[f] + b[f] + e[f] - mu) * inv;
}

__global__ void k_rbfcat(float* rc, const float* rbf, const float* bond) {
  long idx = (long)blockIdx.x * blockDim.x + threadIdx.x;
  if (idx >= (long)NEDGES * RBFCATW) return;
  int e = (int)(idx / RBFCATW), c = (int)(idx % RBFCATW);
  rc[idx] = (c < NRADD) ? rbf[(long)e * NRADD + c]
                        : bond[(long)e * BONDD + (c - NRADD)];
}

__global__ void k_wprod(float* Cw, const float* Aw, const float* Bw,
                        int R, int Kk, int Cc) {
  int i = blockIdx.x * blockDim.x + threadIdx.x;
  if (i >= R * Cc) return;
  int r = i / Cc, c = i % Cc;
  float s = 0.f;
  for (int k = 0; k < Kk; ++k) s += Aw[r * Kk + k] * Bw[k * Cc + c];
  Cw[i] = s;
}

__global__ void k_triplet(float* agg, const float* down, const float* sbuf,
                          const int* ikj, const int* iji) {
  long idx = (long)blockIdx.x * blockDim.x + threadIdx.x;
  if (idx >= (long)NTRIP * INTE) return;
  int t = (int)(idx / INTE), f = (int)(idx % INTE);
  float v = down[(long)ikj[t] * INTE + f] * sbuf[idx];
  atomicAdd(&agg[(long)iji[t] * INTE + f], v);
}

__global__ void k_outedge(float* vnode, const float* rbf, const float* Wr,
                          const float* xe, const int* eidx) {
  long idx = (long)blockIdx.x * blockDim.x + threadIdx.x;
  if (idx >= (long)NEDGES * HIDD) return;
  int e = (int)(idx / HIDD), c = (int)(idx % HIDD);
  const float* re = rbf + (long)e * NRADD;
  float rw = 0.f;
#pragma unroll
  for (int k = 0; k < NRADD; ++k) rw += re[k] * Wr[k * HIDD + c];
  atomicAdd(&vnode[(long)eidx[NEDGES + e] * HIDD + c],
            rw * xe[(long)e * HIDD + c]);
}

__global__ void k_outfinal(float* dout, const float* v, const float* Wout) {
  int n = blockIdx.x * blockDim.x + threadIdx.x;
  if (n >= NNODES) return;
  float s = 0.f;
  const float* row = v + (long)n * OUTE;
  for (int c = 0; c < OUTE; ++c) s += row[c] * Wout[c];
  dout[n] += s;
}

// ---------------------------------------------------------------------------
// host-side orchestration
// ---------------------------------------------------------------------------
static inline void zerok(float* p, long n, hipStream_t s) {
  k_zero<<<dim3((unsigned)((n + 255) / 256)), dim3(256), 0, s>>>(p, n);
}
static inline void gemm(const float* A, const float* W, const float* bias,
                        const float* add, float* C, int M, int N, int K,
                        int act, hipStream_t s) {
  if (N % 128 == 0) {
    dim3 g((unsigned)((M + BM - 1) / BM), (unsigned)(N / 128));
    k_gemm<128><<<g, 256, 0, s>>>(A, W, bias, add, C, M, N, K, act);
  } else {
    dim3 g((unsigned)((M + BM - 1) / BM), (unsigned)((N + 63) / 64));
    k_gemm<64><<<g, 256, 0, s>>>(A, W, bias, add, C, M, N, K, act);
  }
}

struct InterP {
  const float *W, *Wdown, *Wji, *Wkj, *Wrbf1, *Wrbf2, *Wsbf1, *Wsbf2, *Wup;
  const float *aW1, *aW2, *ab1, *ab2, *b, *bW1, *bW2, *bb1, *bb2, *bji, *bkj;
};
struct OutP {
  const float *Wout, *Wrbf, *Wup, *l0W, *l0b, *l1W, *l1b, *l2W, *l2b;
};

extern "C" void kernel_launch(void* const* d_in, const int* in_sizes, int n_in,
                              void* d_out, int out_size, void* d_ws,
                              size_t ws_size, hipStream_t stream) {
  (void)in_sizes; (void)n_in; (void)out_size; (void)ws_size;

  const int*   x_in   = (const int*)d_in[0];
  const int*   eidx   = (const int*)d_in[1];
  const float* dist   = (const float*)d_in[2];
  const float* sbf    = (const float*)d_in[3];
  const int*   idx_kj = (const int*)d_in[4];
  const int*   idx_ji = (const int*)d_in[5];

  // params flattened in sorted-key order (jax tree order)
  const float* atom_emb = (const float*)d_in[6];
  const float* emb_W    = (const float*)d_in[7];
  const float* emb_Wrbf = (const float*)d_in[8];
  const float* emb_b    = (const float*)d_in[9];
  const float* emb_brbf = (const float*)d_in[10];
  const float* freq     = (const float*)d_in[11];
  const float* g_Wgat   = (const float*)d_in[12];
  const float* g_Wout   = (const float*)d_in[13];
  const float* g_Wrbf   = (const float*)d_in[14];
  const float* g_bgat   = (const float*)d_in[15];
  const float* g_bout   = (const float*)d_in[16];
  const float* g_brbf   = (const float*)d_in[17];
  const float* gat_W    = (const float*)d_in[18];
  const float* gat_adst = (const float*)d_in[19];
  const float* gat_asrc = (const float*)d_in[20];
  const float* gat_b    = (const float*)d_in[21];

  InterP ip[NBLK];
  for (int bk = 0; bk < NBLK; ++bk) {
    const float** q = (const float**)&ip[bk];
    for (int t = 0; t < 20; ++t) q[t] = (const float*)d_in[22 + 20 * bk + t];
  }
  OutP op[NBLK + 1];
  for (int ob = 0; ob < NBLK + 1; ++ob) {
    const float** q = (const float**)&op[ob];
    for (int t = 0; t < 9; ++t) q[t] = (const float*)d_in[82 + 9 * ob + t];
  }
  const float* resi_emb = (const float*)d_in[118];

  // workspace layout
  size_t off = 0;
  char* base = (char*)d_ws;
  auto wsf = [&](size_t nelem) -> float* {
    float* p = (float*)(base + off);
    off = (off + nelem * sizeof(float) + 255) & ~(size_t)255;
    return p;
  };
  float* emb    = wsf((size_t)NNODES * HIDD);
  float* rbf    = wsf((size_t)NEDGES * NRADD);
  float* cat    = wsf((size_t)NEDGES * 384);
  float* xe     = wsf((size_t)NEDGES * HIDD);
  float* e1     = wsf((size_t)NEDGES * HIDD);
  float* e2     = wsf((size_t)NEDGES * HIDD);
  float* e3     = wsf((size_t)NEDGES * HIDD);
  float* e4     = wsf((size_t)NEDGES * HIDD);
  float* dwn    = wsf((size_t)NEDGES * INTE);
  float* agg    = wsf((size_t)NEDGES * INTE);
  float* sbuf   = wsf((size_t)NTRIP * INTE);
  float* rbfcat = wsf((size_t)NEDGES * RBFCATW);
  float* bond   = wsf((size_t)NEDGES * BONDD);
  float* wx     = wsf((size_t)NNODES * 512);
  float* gout   = wsf((size_t)NNODES * 512);
  float* xn     = wsf((size_t)NNODES * 512);
  float* xg     = wsf((size_t)NNODES * HIDD);
  float* a_s    = wsf((size_t)NNODES * HEADSN);
  float* a_d    = wsf((size_t)NNODES * HEADSN);
  float* mbuf   = wsf((size_t)NNODES * HEADSN);
  float* den    = wsf((size_t)NNODES * HEADSN);
  float* eatt   = wsf((size_t)(NEDGES + NNODES) * HEADSN);
  float* vnode  = wsf((size_t)NNODES * HIDD);
  float* v1     = wsf((size_t)NNODES * OUTE);
  float* v2     = wsf((size_t)NNODES * OUTE);
  float* wrbf12 = wsf((size_t)RBFCATW * HIDD);
  float* wsbf12 = wsf((size_t)NSR * INTE);

  float* dout = (float*)d_out;
  auto g1 = [](long n) { return dim3((unsigned)((n + 255) / 256)); };

  // ---- node embeddings + radial basis -------------------------------------
  k_node_emb<<<g1((long)NNODES * HIDD), 256, 0, stream>>>(emb, x_in, resi_emb,
                                                          atom_emb);
  k_rbf<<<g1((long)NEDGES * NRADD), 256, 0, stream>>>(rbf, dist, freq);

  // ---- xe = swish([emb_i, emb_j, rh] @ emb_W + emb_b) ---------------------
  k_cat3<<<g1((long)NEDGES * 384), 256, 0, stream>>>(cat, emb, eidx, rbf,
                                                     emb_Wrbf, emb_brbf);
  gemm(cat, emb_W, emb_b, nullptr, xe, NEDGES, HIDD, 384, 1, stream);

  // ---- GAT + InstanceNorm + bond features ---------------------------------
  gemm(emb, gat_W, nullptr, nullptr, wx, NNODES, 512, HIDD, 0, stream);
  k_gat_coef<<<g1((long)NNODES * HEADSN), 256, 0, stream>>>(a_s, a_d, wx,
                                                            gat_asrc, gat_adst);
  k_fill<<<g1((long)NNODES * HEADSN), 256, 0, stream>>>(mbuf, -3.0e38f,
                                                        (long)NNODES * HEADSN);
  zerok(den, (long)NNODES * HEADSN, stream);
  zerok(gout, (long)NNODES * 512, stream);
  k_att1<<<g1((long)(NEDGES + NNODES) * HEADSN), 256, 0, stream>>>(
      eatt, mbuf, a_s, a_d, eidx);
  k_att2<<<g1((long)(NEDGES + NNODES) * HEADSN), 256, 0, stream>>>(eatt, den,
                                                                   mbuf, eidx);
  k_att3<<<g1((long)(NEDGES + NNODES) * 512), 256, 0, stream>>>(gout, eatt,
                                                                den, wx, eidx);
  k_inorm<<<g1((long)NNODES * HEADSN), 256, 0, stream>>>(xn, gout, gat_b, emb);
  gemm(xn, g_Wgat, g_bgat, nullptr, xg, NNODES, HIDD, 512, 1, stream);
  k_cat3<<<g1((long)NEDGES * 384), 256, 0, stream>>>(cat, xg, eidx, rbf,
                                                     g_Wrbf, g_brbf);
  gemm(cat, g_Wout, g_bout, nullptr, bond, NEDGES, BONDD, 384, 1, stream);
  k_rbfcat<<<g1((long)NEDGES * RBFCATW), 256, 0, stream>>>(rbfcat, rbf, bond);

  // ---- output accumulator -------------------------------------------------
  zerok(dout, NNODES, stream);

  auto out_block = [&](const OutP& o) {
    zerok(vnode, (long)NNODES * HIDD, stream);
    k_outedge<<<g1((long)NEDGES * HIDD), 256, 0, stream>>>(vnode, rbf, o.Wrbf,
                                                           xe, eidx);
    gemm(vnode, o.Wup, nullptr, nullptr, v1, NNODES, OUTE, HIDD, 0, stream);
    gemm(v1, o.l0W, o.l0b, nullptr, v2, NNODES, OUTE, OUTE, 1, stream);
    gemm(v2, o.l1W, o.l1b, nullptr, v1, NNODES, OUTE, OUTE, 1, stream);
    gemm(v1, o.l2W, o.l2b, nullptr, v2, NNODES, OUTE, OUTE, 1, stream);
    k_outfinal<<<g1(NNODES), 256, 0, stream>>>(dout, v2, o.Wout);
  };

  out_block(op[0]);

  // ---- interaction blocks -------------------------------------------------
  for (int bk = 0; bk < NBLK; ++bk) {
    const InterP& p = ip[bk];
    // rank-8 weight factor products -> single dense weights
    k_wprod<<<g1((long)RBFCATW * HIDD), 256, 0, stream>>>(wrbf12, p.Wrbf1,
                                                          p.Wrbf2, RBFCATW, 8,
                                                          HIDD);
    k_wprod<<<g1((long)NSR * INTE), 256, 0, stream>>>(wsbf12, p.Wsbf1,
                                                      p.Wsbf2, NSR, 8, INTE);
    gemm(xe, p.Wji, p.bji, nullptr, e1, NEDGES, HIDD, HIDD, 1, stream);
    gemm(xe, p.Wkj, p.bkj, nullptr, e2, NEDGES, HIDD, HIDD, 1, stream);
    gemm(rbfcat, wrbf12, nullptr, nullptr, e3, NEDGES, HIDD, RBFCATW, 0,
         stream);
    k_mul<<<g1((long)NEDGES * HIDD), 256, 0, stream>>>(e2, e2, e3,
                                                       (long)NEDGES * HIDD);
    gemm(e2, p.Wdown, nullptr, nullptr, dwn, NEDGES, INTE, HIDD, 1, stream);
    gemm(sbf, wsbf12, nullptr, nullptr, sbuf, NTRIP, INTE, NSR, 0, stream);
    zerok(agg, (long)NEDGES * INTE, stream);
    k_triplet<<<g1((long)NTRIP * INTE), 256, 0, stream>>>(agg, dwn, sbuf,
                                                          idx_kj, idx_ji);
    // h = x_ji + swish(agg @ Wup)
    gemm(agg, p.Wup, nullptr, e1, e3, NEDGES, HIDD, INTE, 1, stream);
    // before-skip residual
    gemm(e3, p.bW1, p.bb1, nullptr, e2, NEDGES, HIDD, HIDD, 1, stream);
    gemm(e2, p.bW2, p.bb2, e3, e4, NEDGES, HIDD, HIDD, 1, stream);
    // h = swish(h @ W + b) + xe
    gemm(e4, p.W, p.b, xe, e2, NEDGES, HIDD, HIDD, 1, stream);
    // after-skip residual
    gemm(e2, p.aW1, p.ab1, nullptr, e1, NEDGES, HIDD, HIDD, 1, stream);
    gemm(e1, p.aW2, p.ab2, e2, e3, NEDGES, HIDD, HIDD, 1, stream);
    k_copy<<<g1((long)NEDGES * HIDD), 256, 0, stream>>>(xe, e3,
                                                        (long)NEDGES * HIDD);
    out_block(op[bk + 1]);
  }
}
